// ResidualLFQ_50508815401238
// MI455X (gfx1250) — matrix operand order repaired and checked by reading, not compiled, not run on hardware
//
#include <hip/hip_runtime.h>
#include <hip/hip_bf16.h>

// ResidualLFQ forward, fused for gfx1250 (MI455X).
//
// Math notes (all log-base-2 internally; v_exp_f32/v_log_f32 are base-2 ops):
//  - w_d = f16(log2e * 2*scale * r_d); WMMA computes logit2_bk = sum_d w_d * s_kd.
//  - softmax over the 2^12 product code factorizes:
//      lse2_b = sum_d (|w_d| + log2(1 + 2^(-2|w_d|)))
//      per-sample entropy = ln2 * (lse2_b - sum_d w_d * tanh(ln2*w_d))
//        with tanh(ln2*w) = sign(w)*(1-u)/(1+u), u = 2^(-2|w|)
//    -> closed form, no 4096-wide pass needed for the per-sample term.
//  - only avg_p (codebook entropy) needs the 4096-wide mass: p = 2^(logit2 - lse2).
//    Complement symmetry: logit2(~k) = -logit2(k), so one WMMA over columns
//    [0,2048) yields p(col) AND p(4095-col) -> half the WMMAs / B builds.
//    B is built arithmetically from code-index bits (codebook never in memory).

typedef _Float16 v16h __attribute__((ext_vector_type(16)));
typedef float    v8f  __attribute__((ext_vector_type(8)));

#define LFQ_B   8192
#define LFQ_D   12
#define LFQ_K   4096
#define LFQ_NC  4
#define THREADS 256
#define ROWS_PER_BLOCK 128   // 8 waves * 16 rows
#define LOG2E_F 1.44269504f
#define LN2_F   0.69314718f

// raw gfx1250 transcendentals (base-2): v_exp_f32 / v_log_f32
#define EXP2F(x) __builtin_amdgcn_exp2f(x)
#define LOG2F(x) __builtin_amdgcn_logf(x)

// workspace (floats): [0] = sum (r-q)^2 over c,b,d ; [1] = sum per-sample entropy over c,b
//                     [16 .. 16+NC*K) = sum_b p_bk per codebook
#define WS_SCAL  16
#define WS_TOTAL (WS_SCAL + LFQ_NC * LFQ_K)

__global__ void lfq_zero_ws(float* __restrict__ ws) {
  int i = blockIdx.x * blockDim.x + threadIdx.x;
  if (i < WS_TOTAL) ws[i] = 0.0f;
}

__global__ __launch_bounds__(THREADS)
void lfq_main(const float* __restrict__ x, float* __restrict__ out,
              float* __restrict__ ws) {
  __shared__ float s_avgp[LFQ_K];   // 16 KB: block-local sum_b p for current codebook
  __shared__ float s_lse[8 * 16];   // per-wave, per-row base-2 lse
  __shared__ float s_scal[2];       // [0]=commit sqsum, [1]=per-sample entropy sum

  const int tid  = threadIdx.x;
  const int wave = tid >> 5;        // wave32: 8 waves / block
  const int lane = tid & 31;
  const int rowL = lane & 15;       // row within 16x16 tile
  const int hi   = lane >> 4;       // lane half (WMMA layout)
  const int gRow = blockIdx.x * ROWS_PER_BLOCK + wave * 16 + rowL;

  // Load this row of x (rows duplicated across lane halves; needed for WMMA A layout).
  float xr[LFQ_D], r[LFQ_D];
#pragma unroll
  for (int d = 0; d < LFQ_D; ++d) {
    xr[d] = x[gRow * LFQ_D + d];
    r[d]  = xr[d];
  }

  if (tid == 0) { s_scal[0] = 0.0f; s_scal[1] = 0.0f; }
  __syncthreads();

  const float scales[LFQ_NC] = {1.0f, 0.5f, 0.25f, 0.125f};

  for (int c = 0; c < LFQ_NC; ++c) {
    const float sc = scales[c];
    const float s2l = LOG2E_F * 2.0f * sc;   // fold log2e so v_exp_f32 needs no pre-mul

    // Closed-form per-row stats from the *same* f16-rounded w the WMMA consumes.
    float w[LFQ_D], q[LFQ_D];
    float lse2 = 0.0f, wt = 0.0f, csq = 0.0f;
#pragma unroll
    for (int d = 0; d < LFQ_D; ++d) {
      q[d] = (r[d] > 0.0f) ? sc : -sc;
      _Float16 wh = (_Float16)(s2l * r[d]);
      w[d] = (float)wh;
      float aw = fabsf(w[d]);
      float u  = EXP2F(-2.0f * aw);
      lse2 += aw + LOG2F(1.0f + u);           // log2(2^w + 2^-w)
      wt   += aw * (1.0f - u) / (1.0f + u);   // w * tanh(ln2*w)
      float dd = r[d] - q[d];
      csq += dd * dd;
    }
    float ent = LN2_F * (lse2 - wt);          // per-sample entropy (natural log)

    // Zero block-local avg_p; publish lse2; accumulate scalars (one lane-half per row).
    for (int i = tid; i < LFQ_K; i += THREADS) s_avgp[i] = 0.0f;
    if (hi == 0) {
      s_lse[wave * 16 + rowL] = lse2;
      atomicAdd(&s_scal[0], csq);
      atomicAdd(&s_scal[1], ent);
    }
    __syncthreads();

    // lse2 for the 8 rows this lane produces in the D tile (lanes 16-31: rows 8..15).
    float lse8[8];
#pragma unroll
    for (int v = 0; v < 8; ++v) lse8[v] = s_lse[wave * 16 + hi * 8 + v];

    // A fragment (16x32 f16, K=12 used, log2e*2*scale folded in):
    //   lanes 0-15 : elems 0..7 = K0..7 ; lanes 16-31: elems 0..3 = K8..11 (rest 0)
    v16h a;
#pragma unroll
    for (int e = 0; e < 16; ++e) a[e] = (_Float16)0.0f;
    const int cnt  = hi ? 4 : 8;
    const int base = hi ? 8 : 0;
#pragma unroll
    for (int e = 0; e < 8; ++e) {
      int idx = base + e; if (idx > LFQ_D - 1) idx = LFQ_D - 1;
      float wv = (e < cnt) ? w[idx] : 0.0f;
      a[e] = (_Float16)wv;
    }

    // Sweep codes 0..2047 in 16-column tiles; complement gives 2048..4095 for free.
    // B (32x16 f16): column N = tb+lane%16; lanes 0-15 elems 0..7 = +/-1 from code bits 0..7,
    // lanes 16-31 elems 0..3 = bits 8..11 (bits >=4 of col are wave-uniform -> SALU); rest 0.
    for (int tb = 0; tb < LFQ_K / 2; tb += 16) {
      const int col = tb + rowL;
      v16h b;
#pragma unroll
      for (int e = 0; e < 16; ++e) {
        float sv = 0.0f;
        if (e < (hi ? 4 : 8)) sv = ((col >> (8 * hi + e)) & 1) ? 1.0f : -1.0f;
        b[e] = (_Float16)sv;
      }
      v8f acc = {};
      acc = __builtin_amdgcn_wmma_f32_16x16x32_f16(
          /*neg_a=*/false, a, /*neg_b=*/false, b,
          /*c_mod=*/(short)0, acc, /*reuse_a=*/false, /*reuse_b=*/false);

      // p(col) = 2^(l - lse2), p(4095-col) = 2^(-l - lse2); fold 8 rows per lane.
      float colsumP = 0.0f, colsumN = 0.0f;
#pragma unroll
      for (int v = 0; v < 8; ++v) {
        colsumP += EXP2F( acc[v] - lse8[v]);
        colsumN += EXP2F(-acc[v] - lse8[v]);
      }
      atomicAdd(&s_avgp[col],        colsumP);  // ds_add_f32
      atomicAdd(&s_avgp[4095 - col], colsumN);
    }
    __syncthreads();

    // Flush block-local avg_p into global accumulator for this codebook.
    for (int i = tid; i < LFQ_K; i += THREADS)
      atomicAdd(&ws[WS_SCAL + c * LFQ_K + i], s_avgp[i]);

    // Residual update.
#pragma unroll
    for (int d = 0; d < LFQ_D; ++d) r[d] -= q[d];

    __syncthreads();   // protect s_avgp / s_lse for next codebook
  }

  // xh = x - final residual (straight-through sum of q's).
  if (hi == 0) {
#pragma unroll
    for (int d = 0; d < LFQ_D; ++d) out[gRow * LFQ_D + d] = xr[d] - r[d];
  }
  if (tid == 0) {
    atomicAdd(&ws[0], s_scal[0]);
    atomicAdd(&ws[1], s_scal[1]);
  }
}

__global__ __launch_bounds__(THREADS)
void lfq_finalize(const float* __restrict__ ws, float* __restrict__ out) {
  __shared__ float red[THREADS];
  const int tid = threadIdx.x;

  // codebook entropy summed over all 4 codebooks: -sum avg_p * ln(avg_p + eps)
  float local = 0.0f;
  for (int i = tid; i < LFQ_NC * LFQ_K; i += THREADS) {
    float ap = ws[WS_SCAL + i] * (1.0f / (float)LFQ_B);
    local += -ap * (LN2_F * LOG2F(ap + 1e-9f));   // ln = ln2 * log2
  }
  red[tid] = local;
  __syncthreads();
  for (int s = THREADS / 2; s > 0; s >>= 1) {
    if (tid < s) red[tid] += red[tid + s];
    __syncthreads();
  }

  if (tid == 0) {
    float cb_ent  = red[0];
    float commit  = ws[0] * (1.0f / (float)(LFQ_B * LFQ_D));  // == codebook loss
    float ps_ent  = ws[1] * (1.0f / (float)LFQ_B);
    float entloss = ps_ent - cb_ent;
    float total   = 0.1f * commit + 0.1f * commit + entloss;
    out[LFQ_B * LFQ_D + 0] = total;
    out[LFQ_B * LFQ_D + 1] = commit;
    out[LFQ_B * LFQ_D + 2] = commit;
    out[LFQ_B * LFQ_D + 3] = entloss;
  }
}

extern "C" void kernel_launch(void* const* d_in, const int* in_sizes, int n_in,
                              void* d_out, int out_size, void* d_ws, size_t ws_size,
                              hipStream_t stream) {
  (void)in_sizes; (void)n_in; (void)out_size; (void)ws_size;
  const float* x  = (const float*)d_in[0];
  float* out      = (float*)d_out;
  float* ws       = (float*)d_ws;

  lfq_zero_ws<<<(WS_TOTAL + THREADS - 1) / THREADS, THREADS, 0, stream>>>(ws);
  lfq_main<<<LFQ_B / ROWS_PER_BLOCK, THREADS, 0, stream>>>(x, out, ws);
  lfq_finalize<<<1, THREADS, 0, stream>>>(ws, out);
}
